// GraphCNN_Generalization_36636071035353
// MI455X (gfx1250) — matrix operand order repaired and checked by reading, not compiled, not run on hardware
//
#include <hip/hip_runtime.h>
#include <hip/hip_bf16.h>

// ---------------------------------------------------------------------------
// Problem constants (from the reference)
// ---------------------------------------------------------------------------
constexpr int Nn    = 12288;   // nodes
constexpr int IN_C  = 512;
constexpr int HID_C = 256;
constexpr int OUT_C = 8;
constexpr int Ee    = 196608;  // edges (before self loops)
constexpr int ET    = Ee + Nn; // edges incl self loops

typedef __attribute__((ext_vector_type(16))) __bf16 v16bf;
typedef __attribute__((ext_vector_type(8)))  float  v8f;
typedef __attribute__((ext_vector_type(8)))  int    v8i;

// ---------------------------------------------------------------------------
// helpers
// ---------------------------------------------------------------------------
__device__ __forceinline__ unsigned short f2bf(float f) {
    unsigned u = __float_as_uint(f);
    u = (u + 0x7fffu + ((u >> 16) & 1u)) >> 16;   // round-to-nearest-even
    return (unsigned short)u;
}

__device__ __forceinline__ void atomicMaxF(float* addr, float val) {
    unsigned int* ua = (unsigned int*)addr;
    unsigned int old = *((volatile unsigned int*)ua);
    while (__uint_as_float(old) < val) {
        unsigned int assumed = old;
        old = atomicCAS(ua, assumed, __float_as_uint(val));
        if (old == assumed) break;
    }
}

// CDNA5 async global -> LDS DMA (ASYNCcnt-tracked), ISA 08_async_tensor §4.
__device__ __forceinline__ void async_b128_to_lds(unsigned lds_off,
                                                  unsigned long long gaddr) {
    asm volatile("global_load_async_to_lds_b128 %0, %1, off"
                 :: "v"(lds_off), "v"(gaddr) : "memory");
}

// ---------------------------------------------------------------------------
// generic bf16 WMMA GEMM:  C[M x P] = A[M x K] * B[P x K]^T  (both row-major
// over K), optional per-column scale  C[:,j] *= colscale[j].
//
// Block = 256 threads (8 wave32s). Block tile 64(M) x 64(P); K step 64.
// Waves: wi = wid&3 -> 16-row strip; wj = wid>>2 -> 32-col strip (two 16x16
// accumulators with A-fragment reuse). Staging: double-buffered LDS filled by
// GLOBAL_LOAD_ASYNC_TO_LDS_B128, synced with s_wait_asynccnt, so the DMA of
// K-slice s+1 overlaps the 4 WMMAs of slice s.
// ---------------------------------------------------------------------------
__global__ __launch_bounds__(256)
void wmma_gemm_nt(const unsigned short* __restrict__ A,
                  const unsigned short* __restrict__ B,
                  float* __restrict__ C,
                  int K, int ldc,
                  const float* __restrict__ colscale)
{
    __shared__ __align__(16) unsigned short As[2][64 * 64];
    __shared__ __align__(16) unsigned short Bs[2][64 * 64];

    const int tid  = threadIdx.x;
    const int wid  = tid >> 5;
    const int lane = tid & 31;
    const int iblk = blockIdx.y * 64;
    const int jblk = blockIdx.x * 64;
    const int wi   = wid & 3;   // row sub-tile (0..3)
    const int wj   = wid >> 2;  // col strip   (0..1), 32 cols each

    const int ln16    = lane & 15;
    const int khalf_a = (lane < 16) ? 0 : 8;
    const int khalf_b = (lane < 16) ? 0 : 16;

    // cooperative DMA geometry: 64 rows x 128B per tile; each thread moves
    // 2 x 16B for A and 2 x 16B for B per K step.
    const int lrow = tid >> 2;          // 0..63
    const int le0  = (tid & 3) * 16;    // bf16 element offset (32B per thread)

    const unsigned long long aBase =
        (unsigned long long)(const void*)A + ((size_t)(iblk + lrow) * K + le0) * 2;
    const unsigned long long bBase =
        (unsigned long long)(const void*)B + ((size_t)(jblk + lrow) * K + le0) * 2;
    const unsigned aOff0 = (unsigned)(size_t)&As[0][lrow * 64 + le0];
    const unsigned bOff0 = (unsigned)(size_t)&Bs[0][lrow * 64 + le0];
    const unsigned aOff1 = (unsigned)(size_t)&As[1][lrow * 64 + le0];
    const unsigned bOff1 = (unsigned)(size_t)&Bs[1][lrow * 64 + le0];

    auto issue = [&](int buf, int k0) {
        unsigned long long ga = aBase + (unsigned long long)k0 * 2;
        unsigned long long gb = bBase + (unsigned long long)k0 * 2;
        unsigned ao = buf ? aOff1 : aOff0;
        unsigned bo = buf ? bOff1 : bOff0;
        async_b128_to_lds(ao,      ga);
        async_b128_to_lds(ao + 16, ga + 16);
        async_b128_to_lds(bo,      gb);
        async_b128_to_lds(bo + 16, gb + 16);
    };

    v8f acc0 = {}, acc1 = {};
    const int steps = K >> 6;

    issue(0, 0);
    for (int s = 0; s < steps; ++s) {
        const int cur = s & 1;
        if (s + 1 < steps) {
            issue(cur ^ 1, (s + 1) << 6);
            asm volatile("s_wait_asynccnt 0x4" ::: "memory");  // batch s landed
        } else {
            asm volatile("s_wait_asynccnt 0x0" ::: "memory");
        }
        __syncthreads();

        const unsigned short* ar  = &As[cur][(wi * 16 + ln16) * 64];
        const unsigned short* br0 = &Bs[cur][(wj * 32 + ln16) * 64];
        const unsigned short* br1 = br0 + 16 * 64;

        #pragma unroll
        for (int kk = 0; kk < 64; kk += 32) {
            union { v8i i; v16bf b; } fa, fb0, fb1;
            #pragma unroll
            for (int v = 0; v < 8; ++v) {
                int ka = kk + khalf_b + 2 * v;
                fa.i[v]  = *(const unsigned int*)(ar + ka);     // ds_load_b32/b128
                int kb = kk + khalf_b + 2 * v;
                fb0.i[v] = *(const unsigned int*)(br0 + kb);
                fb1.i[v] = *(const unsigned int*)(br1 + kb);
            }
            acc0 = __builtin_amdgcn_wmma_f32_16x16x32_bf16(
                       false, fa.b, false, fb0.b, (short)0, acc0, false, false);
            acc1 = __builtin_amdgcn_wmma_f32_16x16x32_bf16(
                       false, fa.b, false, fb1.b, (short)0, acc1, false, false);
        }
        __syncthreads();   // protect buffer reuse two steps ahead
    }

    // ---- epilogue: VGPR g -> rows {g, 8+g}; cols = strip base + lane&15 ----
    const int col0  = jblk + wj * 32 + ln16;
    const int col1  = col0 + 16;
    const int rbase = iblk + wi * 16 + ((lane < 16) ? 0 : 8);
    float cs0 = colscale ? colscale[col0] : 1.0f;
    float cs1 = colscale ? colscale[col1] : 1.0f;
    #pragma unroll
    for (int g = 0; g < 8; ++g) {
        C[(size_t)(rbase + g) * (size_t)ldc + col0] = acc0[g] * cs0;
        C[(size_t)(rbase + g) * (size_t)ldc + col1] = acc1[g] * cs1;
    }
}

// ---------------------------------------------------------------------------
// small elementwise / reduction kernels
// ---------------------------------------------------------------------------
__global__ void prep_scale_k(const float* __restrict__ emw,
                             const float* __restrict__ lgp,
                             float* __restrict__ scale)
{
    int c = blockIdx.x * blockDim.x + threadIdx.x;
    if (c >= IN_C) return;
    const float EPSf = 2.220446049250313e-16f;
    float p      = 1.f / (1.f + expf(-lgp[c]));
    float approx = logf(p + EPSf) - logf(1.f - p + EPSf);  // log(0.5) terms cancel
    float keep   = 1.f - 1.f / (1.f + expf(-approx / 0.1f));
    float w      = 1.f / (1.f + expf(-emw[c]));
    scale[c] = keep * w;
}

__global__ __launch_bounds__(256)
void make_xe_k(const float* __restrict__ x, const float* __restrict__ scale,
               unsigned short* __restrict__ xe, unsigned short* __restrict__ xbf,
               float* __restrict__ rowsum)
{
    __shared__ float sr[256];
    const int i = blockIdx.x, t = threadIdx.x;
    float part = 0.f;
    for (int c = t; c < IN_C; c += 256) {
        float xv = x[(size_t)i * IN_C + c];
        float v  = xv * scale[c];
        xe [(size_t)i * IN_C + c] = f2bf(v);
        xbf[(size_t)i * IN_C + c] = f2bf(xv);
        part += v;
    }
    sr[t] = part; __syncthreads();
    for (int o = 128; o > 0; o >>= 1) { if (t < o) sr[t] += sr[t + o]; __syncthreads(); }
    if (t == 0) rowsum[i] = sr[0];
}

__global__ void inv_rowsum_k(const float* __restrict__ rs, float* __restrict__ inv, int n)
{
    int i = blockIdx.x * blockDim.x + threadIdx.x;
    if (i < n) inv[i] = 1.f / (rs[i] + 1e-6f);
}

__global__ void w1t_k(const float* __restrict__ W1, unsigned short* __restrict__ W1t)
{
    int idx = blockIdx.x * blockDim.x + threadIdx.x;   // IN_C * HID_C
    if (idx >= IN_C * HID_C) return;
    int k = idx / HID_C, p = idx - k * HID_C;
    W1t[(size_t)p * IN_C + k] = f2bf(W1[idx]);
}

__global__ void fill_f32_k(float* __restrict__ p, float v, int n)
{
    int i = blockIdx.x * blockDim.x + threadIdx.x;
    if (i < n) p[i] = v;
}

__global__ __launch_bounds__(256)
void alpha256_k(const float* __restrict__ xp, const float* __restrict__ a_s,
                const float* __restrict__ a_d, float* __restrict__ as,
                float* __restrict__ ad)
{
    __shared__ float s0[256], s1[256];
    const int i = blockIdx.x, t = threadIdx.x;
    float v = xp[(size_t)i * HID_C + t];
    s0[t] = v * a_s[t]; s1[t] = v * a_d[t];
    __syncthreads();
    for (int o = 128; o > 0; o >>= 1) {
        if (t < o) { s0[t] += s0[t + o]; s1[t] += s1[t + o]; }
        __syncthreads();
    }
    if (t == 0) { as[i] = s0[0]; ad[i] = s1[0]; }
}

__global__ void alpha8_k(const float* __restrict__ xp, const float* __restrict__ a_s,
                         const float* __restrict__ a_d, float* __restrict__ as,
                         float* __restrict__ ad)
{
    int i = blockIdx.x * blockDim.x + threadIdx.x;
    if (i >= Nn) return;
    float s = 0.f, t = 0.f;
    #pragma unroll
    for (int o = 0; o < OUT_C; ++o) {
        float v = xp[(size_t)i * OUT_C + o];
        s += v * a_s[o]; t += v * a_d[o];
    }
    as[i] = s; ad[i] = t;
}

__device__ __forceinline__ void edge_sd(int e, const int* ei, int& s, int& d)
{
    if (e < Ee) { s = ei[e]; d = ei[Ee + e]; }   // edge_index[0,:], edge_index[1,:]
    else        { s = d = e - Ee; }              // self loop
}

__global__ void edge_lrelu_max_k(const int* __restrict__ ei,
                                 const float* __restrict__ as,
                                 const float* __restrict__ ad,
                                 float* __restrict__ elog, float* __restrict__ m)
{
    int e = blockIdx.x * blockDim.x + threadIdx.x;
    if (e >= ET) return;
    int s, d; edge_sd(e, ei, s, d);
    float l = as[s] + ad[d];
    l = (l > 0.f) ? l : 0.2f * l;          // GAT leaky-relu, slope 0.2
    elog[e] = l;
    atomicMaxF(&m[d], l);
}

__global__ void edge_exp_sum_k(const int* __restrict__ ei,
                               float* __restrict__ elog,
                               const float* __restrict__ m,
                               float* __restrict__ den)
{
    int e = blockIdx.x * blockDim.x + threadIdx.x;
    if (e >= ET) return;
    int s, d; edge_sd(e, ei, s, d);
    float ex = expf(elog[e] - m[d]);
    elog[e] = ex;
    atomicAdd(&den[d], ex);
}

__global__ void edge_agg_k(const int* __restrict__ ei,
                           const float* __restrict__ ex,
                           const float* __restrict__ den,
                           const float* __restrict__ xp,
                           float* __restrict__ acc, int cshift)
{
    long idx = (long)blockIdx.x * blockDim.x + threadIdx.x;
    int  C   = 1 << cshift;
    long e   = idx >> cshift;
    int  c   = (int)(idx & (C - 1));
    if (e >= ET) return;
    int s, d; edge_sd((int)e, ei, s, d);
    float w = ex[e] / (den[d] + 1e-16f);
    atomicAdd(&acc[(size_t)d * C + c], w * xp[(size_t)s * C + c]);
}

__global__ void post1_k(float* __restrict__ acc, const float* __restrict__ b1,
                        const float* __restrict__ g, const float* __restrict__ be,
                        const float* __restrict__ mu, const float* __restrict__ var)
{
    int idx = blockIdx.x * blockDim.x + threadIdx.x;  // Nn * HID_C threads
    int c = idx & (HID_C - 1);
    float v = acc[idx] + b1[c];
    v = (v - mu[c]) * rsqrtf(var[c] + 1e-5f) * g[c] + be[c];
    acc[idx] = (v > 0.f) ? v : 0.01f * v;             // torch leaky-relu 0.01
}

__global__ void gemm_hw2_k(const float* __restrict__ hid,
                           const float* __restrict__ W2, float* __restrict__ xp2)
{
    int idx = blockIdx.x * blockDim.x + threadIdx.x;  // Nn * OUT_C
    if (idx >= Nn * OUT_C) return;
    int i = idx >> 3, o = idx & 7;
    float s = 0.f;
    for (int c = 0; c < HID_C; ++c) s += hid[(size_t)i * HID_C + c] * W2[c * OUT_C + o];
    xp2[idx] = s;
}

__global__ void post2_k(const float* __restrict__ acc2, const float* __restrict__ b2,
                        float* __restrict__ out_log, float* __restrict__ node_p)
{
    int i = blockIdx.x * blockDim.x + threadIdx.x;
    if (i >= Nn) return;
    float l[OUT_C], mx = -3.402823466e38f;
    #pragma unroll
    for (int o = 0; o < OUT_C; ++o) {
        l[o] = acc2[(size_t)i * OUT_C + o] + b2[o];
        mx = fmaxf(mx, l[o]);
    }
    float s = 0.f;
    #pragma unroll
    for (int o = 0; o < OUT_C; ++o) s += expf(l[o] - mx);
    float lse = logf(s);
    #pragma unroll
    for (int o = 0; o < OUT_C; ++o) out_log[(size_t)i * OUT_C + o] = l[o] - mx - lse;
    node_p[i] = expf(out_log[(size_t)i * OUT_C + 1]);
}

__global__ __launch_bounds__(256)
void loss_k(const float* __restrict__ out_log, const int* __restrict__ y,
            const int* __restrict__ mask, float* __restrict__ loss)
{
    __shared__ float sa[256], sb[256];
    const int t = threadIdx.x;
    float pa = 0.f, pb = 0.f;
    for (int i = t; i < Nn; i += 256) {
        float mk = (float)mask[i];
        pa += mk * out_log[(size_t)i * OUT_C + y[i]];
        pb += mk;
    }
    sa[t] = pa; sb[t] = pb; __syncthreads();
    for (int o = 128; o > 0; o >>= 1) {
        if (t < o) { sa[t] += sa[t + o]; sb[t] += sb[t + o]; }
        __syncthreads();
    }
    if (t == 0) loss[0] = -sa[0] / sb[0];
}

__global__ __launch_bounds__(256)
void final_out1_k(const float* __restrict__ graph, const float* __restrict__ node_p,
                  float* __restrict__ out1)
{
    __shared__ float s0[256], s1[256];
    const int i = blockIdx.x, t = threadIdx.x;
    float a0 = 0.f, a1 = 0.f;
    for (int j = t; j < Nn; j += 256) {
        float g = graph[(size_t)i * Nn + j];
        float p = node_p[j];
        a1 += g * p;
        a0 += g * (1.f - p);
    }
    s0[t] = a0; s1[t] = a1; __syncthreads();
    for (int o = 128; o > 0; o >>= 1) {
        if (t < o) { s0[t] += s0[t + o]; s1[t] += s1[t + o]; }
        __syncthreads();
    }
    if (t == 0) {
        float v0 = s0[0], v1 = s1[0];
        float mx = fmaxf(v0, v1);
        float e0 = expf(v0 - mx), e1 = expf(v1 - mx);
        float inv = 1.f / (e0 + e1);
        out1[(size_t)i * 2 + 0] = e0 * inv;   // softmax([adj_p_nag, adj_p])
        out1[(size_t)i * 2 + 1] = e1 * inv;
    }
}

// ---------------------------------------------------------------------------
// launch
// ---------------------------------------------------------------------------
extern "C" void kernel_launch(void* const* d_in, const int* in_sizes, int n_in,
                              void* d_out, int out_size, void* d_ws, size_t ws_size,
                              hipStream_t stream)
{
    const float* x    = (const float*)d_in[0];
    const int*   ei   = (const int*)  d_in[1];
    const int*   y    = (const int*)  d_in[2];
    const int*   mask = (const int*)  d_in[3];
    const float* W1   = (const float*)d_in[4];
    const float* as1v = (const float*)d_in[5];
    const float* ad1v = (const float*)d_in[6];
    const float* b1   = (const float*)d_in[7];
    const float* W2   = (const float*)d_in[8];
    const float* as2v = (const float*)d_in[9];
    const float* ad2v = (const float*)d_in[10];
    const float* b2   = (const float*)d_in[11];
    const float* bng  = (const float*)d_in[12];
    const float* bnb  = (const float*)d_in[13];
    const float* bnm  = (const float*)d_in[14];
    const float* bnv  = (const float*)d_in[15];
    const float* emw  = (const float*)d_in[16];
    const float* lgp  = (const float*)d_in[17];

    // outputs: out1 [N,2] | loss [1] | graph [N,N]
    float* out   = (float*)d_out;
    float* out1  = out;
    float* loss  = out + (size_t)2 * Nn;
    float* graph = out + (size_t)2 * Nn + 1;

    // workspace carve-out
    char*  ws  = (char*)d_ws;
    size_t off = 0;
    auto alloc = [&](size_t bytes) -> void* {
        void* p = ws + off;
        off = (off + bytes + 255) & ~(size_t)255;
        return p;
    };
    unsigned short* xe   = (unsigned short*)alloc((size_t)Nn * IN_C * 2);
    unsigned short* xbf  = (unsigned short*)alloc((size_t)Nn * IN_C * 2);
    unsigned short* W1t  = (unsigned short*)alloc((size_t)IN_C * HID_C * 2);
    float* xp1    = (float*)alloc((size_t)Nn * HID_C * 4);
    float* acc1   = (float*)alloc((size_t)Nn * HID_C * 4);   // becomes hid in-place
    float* scale  = (float*)alloc(IN_C * 4);
    float* rowsum = (float*)alloc(Nn * 4);
    float* invrs  = (float*)alloc(Nn * 4);
    float* as1    = (float*)alloc(Nn * 4);
    float* ad1    = (float*)alloc(Nn * 4);
    float* m1     = (float*)alloc(Nn * 4);
    float* den1   = (float*)alloc(Nn * 4);
    float* elog   = (float*)alloc((size_t)ET * 4);           // reused for layer 2
    float* xp2    = (float*)alloc((size_t)Nn * OUT_C * 4);
    float* as2    = (float*)alloc(Nn * 4);
    float* ad2    = (float*)alloc(Nn * 4);
    float* m2     = (float*)alloc(Nn * 4);
    float* den2   = (float*)alloc(Nn * 4);
    float* acc2   = (float*)alloc((size_t)Nn * OUT_C * 4);
    float* outlog = (float*)alloc((size_t)Nn * OUT_C * 4);
    float* nodep  = (float*)alloc(Nn * 4);

    const float NEG_INF = -3.402823466e38f;

    // ---- feature prep ----
    prep_scale_k<<<(IN_C + 255) / 256, 256, 0, stream>>>(emw, lgp, scale);
    make_xe_k<<<Nn, 256, 0, stream>>>(x, scale, xe, xbf, rowsum);
    inv_rowsum_k<<<(Nn + 255) / 256, 256, 0, stream>>>(rowsum, invrs, Nn);
    w1t_k<<<(IN_C * HID_C + 255) / 256, 256, 0, stream>>>(W1, W1t);

    // ---- GAT layer 1: xp1 = x @ W1 on WMMA ----
    wmma_gemm_nt<<<dim3(HID_C / 64, Nn / 64), 256, 0, stream>>>(
        xbf, W1t, xp1, IN_C, HID_C, nullptr);
    alpha256_k<<<Nn, 256, 0, stream>>>(xp1, as1v, ad1v, as1, ad1);

    fill_f32_k<<<(Nn + 255) / 256, 256, 0, stream>>>(m1, NEG_INF, Nn);
    fill_f32_k<<<(Nn + 255) / 256, 256, 0, stream>>>(den1, 0.f, Nn);
    fill_f32_k<<<(Nn * HID_C + 255) / 256, 256, 0, stream>>>(acc1, 0.f, Nn * HID_C);

    edge_lrelu_max_k<<<(ET + 255) / 256, 256, 0, stream>>>(ei, as1, ad1, elog, m1);
    edge_exp_sum_k  <<<(ET + 255) / 256, 256, 0, stream>>>(ei, elog, m1, den1);
    edge_agg_k<<<(int)(((long)ET * HID_C + 255) / 256), 256, 0, stream>>>(
        ei, elog, den1, xp1, acc1, 8 /* C=256 */);
    post1_k<<<(Nn * HID_C + 255) / 256, 256, 0, stream>>>(acc1, b1, bng, bnb, bnm, bnv);

    // ---- GAT layer 2 (tiny GEMM, C=8) ----
    gemm_hw2_k<<<(Nn * OUT_C + 255) / 256, 256, 0, stream>>>(acc1, W2, xp2);
    alpha8_k<<<(Nn + 255) / 256, 256, 0, stream>>>(xp2, as2v, ad2v, as2, ad2);

    fill_f32_k<<<(Nn + 255) / 256, 256, 0, stream>>>(m2, NEG_INF, Nn);
    fill_f32_k<<<(Nn + 255) / 256, 256, 0, stream>>>(den2, 0.f, Nn);
    fill_f32_k<<<(Nn * OUT_C + 255) / 256, 256, 0, stream>>>(acc2, 0.f, Nn * OUT_C);

    edge_lrelu_max_k<<<(ET + 255) / 256, 256, 0, stream>>>(ei, as2, ad2, elog, m2);
    edge_exp_sum_k  <<<(ET + 255) / 256, 256, 0, stream>>>(ei, elog, m2, den2);
    edge_agg_k<<<(int)(((long)ET * OUT_C + 255) / 256), 256, 0, stream>>>(
        ei, elog, den2, xp2, acc2, 3 /* C=8 */);
    post2_k<<<(Nn + 255) / 256, 256, 0, stream>>>(acc2, b2, outlog, nodep);
    loss_k<<<1, 256, 0, stream>>>(outlog, y, mask, loss);

    // ---- Gram matrix on WMMA: graph = (xe @ xe^T) * inv(rowsum_col) ----
    wmma_gemm_nt<<<dim3(Nn / 64, Nn / 64), 256, 0, stream>>>(
        xe, xe, graph, IN_C, Nn, invrs);

    // ---- out1 = softmax([graph@(1-p), graph@p]) ----
    final_out1_k<<<Nn, 256, 0, stream>>>(graph, nodep, out1);
}
